// TransformerBlock_46686294507948
// MI455X (gfx1250) — matrix operand order repaired and checked by reading, not compiled, not run on hardware
//
#include <hip/hip_runtime.h>
#include <hip/hip_bf16.h>
#include <math.h>

typedef unsigned short u16;
typedef unsigned int   u32;
typedef unsigned long long u64;
typedef __attribute__((ext_vector_type(16))) __bf16 v16bf;
typedef __attribute__((ext_vector_type(8)))  float  v8f;
typedef __attribute__((ext_vector_type(4))) unsigned int u32x4;
typedef __attribute__((ext_vector_type(8))) int i32x8;
typedef __attribute__((ext_vector_type(4))) int i32x4;

union Frag16 { uint4 u[2]; v16bf v; };
union F8     { v8f v; float f[8]; };

__device__ __forceinline__ u16 f2bf(float f) {
  u32 u = __float_as_uint(f);
  u32 r = u + 0x7fffu + ((u >> 16) & 1u);   // round-to-nearest-even
  return (u16)(r >> 16);
}

// ---------------------------------------------------------------------------
// TDM: issue a 2D tensor tile load (rows x 64 bf16) into LDS with +8-element
// row padding (pad 4 DWORDs after every 32 DWORDs -> LDS row stride 72 elems).
// D# packing per CDNA5 ISA ch.8 (group0: flags/lds/global, group1: dims).
// ---------------------------------------------------------------------------
__device__ __forceinline__ void tdm_load_tile_2d(
    u32 lds_byte_off, const void* gptr,
    u32 tensor_d0,            // row length in elements (K)
    u32 tensor_d1,            // number of rows in tensor (M or N)
    u32 tile_d0,              // tile row length in elements (64)
    u32 tile_d1,              // tile rows (128)
    u64 stride_elems)         // elements between consecutive rows (K)
{
  u64 ga = (u64)(uintptr_t)gptr;
  u32x4 g0;
  g0[0] = 1u;                                  // count=1, is_restore=0
  g0[1] = lds_byte_off;                        // lds_addr
  g0[2] = (u32)ga;                             // global_addr[31:0]
  g0[3] = (u32)(ga >> 32) | (2u << 30);        // global_addr[56:32] | type=2

  i32x8 g1;
  g1[0] = (int)((1u << 16)                     // data_size = 2 bytes
              | (1u << 20)                     // pad_enable
              | (4u << 22)                     // pad_interval: 32 DWORDs
              | (3u << 25));                   // pad_amount: 4 DWORDs (8 elems)
  g1[1] = (int)((tensor_d0 & 0xffffu) << 16);  // tensor_dim0[15:0] @ bits 63:48
  g1[2] = (int)((tensor_d0 >> 16) | ((tensor_d1 & 0xffffu) << 16));
  g1[3] = (int)((tensor_d1 >> 16) | (tile_d0 << 16));   // tile_dim0 @ 127:112
  g1[4] = (int)(tile_d1);                      // tile_dim1; tile_dim2=0
  g1[5] = (int)(u32)stride_elems;              // tensor_dim0_stride[31:0]
  g1[6] = (int)(u32)(stride_elems >> 32);      // stride[47:32]; dim1_stride lo=0
  g1[7] = 0;

  i32x4 gz = {0, 0, 0, 0};
#if defined(__clang_major__) && (__clang_major__ >= 23)
  i32x8 gz8 = {0, 0, 0, 0, 0, 0, 0, 0};
  __builtin_amdgcn_tensor_load_to_lds(g0, g1, gz, gz, gz8, 0);
#else
  __builtin_amdgcn_tensor_load_to_lds(g0, g1, gz, gz, 0);
#endif
}

// ---------------------------------------------------------------------------
// Weight convert + transpose: in fp32 [R][C] -> out bf16 [C][R]
// ---------------------------------------------------------------------------
__global__ __launch_bounds__(256) void convert_transpose_kernel(
    const float* __restrict__ in, u16* __restrict__ out, int R, int C) {
  size_t idx = (size_t)blockIdx.x * 256 + threadIdx.x;
  size_t total = (size_t)R * C;
  if (idx >= total) return;
  int r = (int)(idx / C);
  int c = (int)(idx % C);
  out[(size_t)c * R + r] = f2bf(in[idx]);
}

// ---------------------------------------------------------------------------
// LayerNorm (dim=1024) -> bf16. One block per row, 256 threads x 4 elems.
// ---------------------------------------------------------------------------
__global__ __launch_bounds__(256) void ln_bf16_kernel(
    const float* __restrict__ x, const float* __restrict__ g,
    const float* __restrict__ b, u16* __restrict__ out) {
  __shared__ float red[16];
  const int row = blockIdx.x;
  const int tid = threadIdx.x;
  const float* xr = x + (size_t)row * 1024;
  float4 v = *(const float4*)(xr + tid * 4);
  float s  = v.x + v.y + v.z + v.w;
  float ss = v.x*v.x + v.y*v.y + v.z*v.z + v.w*v.w;
  #pragma unroll
  for (int off = 16; off > 0; off >>= 1) {
    s  += __shfl_xor(s,  off, 32);
    ss += __shfl_xor(ss, off, 32);
  }
  const int wid = tid >> 5;
  if ((tid & 31) == 0) { red[wid] = s; red[8 + wid] = ss; }
  __syncthreads();
  if (tid < 32) {
    float a = (tid < 8) ? red[tid]     : 0.f;
    float c = (tid < 8) ? red[8 + tid] : 0.f;
    #pragma unroll
    for (int off = 4; off > 0; off >>= 1) {
      a += __shfl_xor(a, off, 32);
      c += __shfl_xor(c, off, 32);
    }
    if (tid == 0) { red[0] = a; red[1] = c; }
  }
  __syncthreads();
  const float mean = red[0] * (1.0f / 1024.0f);
  const float var  = red[1] * (1.0f / 1024.0f) - mean * mean;
  const float inv  = rsqrtf(var + 1e-5f);
  #pragma unroll
  for (int j = 0; j < 4; ++j) {
    int c = tid * 4 + j;
    float xv = (j == 0) ? v.x : (j == 1) ? v.y : (j == 2) ? v.z : v.w;
    float h = (xv - mean) * inv * g[c] + b[c];
    out[(size_t)row * 1024 + c] = f2bf(h);
  }
}

// ---------------------------------------------------------------------------
// Tiled bf16 WMMA GEMM: C[M,N] = A[M,K] * Bt[N,K]^T
// Workgroup tile 128x128, BK=64; 8 waves (2x4), wave tile 64x32 = 8 accum
// frags. A/B tiles staged by the Tensor Data Mover, double-buffered so the
// DMA for tile k+1 overlaps WMMA on tile k.
// Epilogues: 0 = QKV head-scatter (bf16), 1 = +bias +resid -> fp32,
//            2 = +bias, GELU -> bf16
// ---------------------------------------------------------------------------
#define BK     64
#define LDS_A  72   // 64 + 8 pad (row stride 144B, 16B aligned)

template <int EPI>
__global__ __launch_bounds__(256) void gemm_bf16_wmma(
    const u16* __restrict__ A, const u16* __restrict__ Bt,
    int M, int N, int K,
    const float* __restrict__ bias, const float* __restrict__ resid,
    float* __restrict__ outF, u16* __restrict__ outB,
    u16* __restrict__ qout, u16* __restrict__ kout, u16* __restrict__ vout) {
  __shared__ u16 As[2][128 * LDS_A];
  __shared__ u16 Bs[2][128 * LDS_A];

  const int tid  = threadIdx.x;
  const int lane = tid & 31;
  const int wid  = tid >> 5;
  const int wm   = wid >> 2;     // 0..1 : 64-row wave tile
  const int wn   = wid & 3;      // 0..3 : 32-col wave tile
  const int half = lane >> 4;
  const int lr   = lane & 15;

  const int n0 = blockIdx.x * 128;
  const int m0 = blockIdx.y * 128;
  const int nk = K / BK;

  F8 acc[4][2];
  #pragma unroll
  for (int i = 0; i < 4; ++i)
    #pragma unroll
    for (int j = 0; j < 2; ++j)
      #pragma unroll
      for (int r = 0; r < 8; ++r) acc[i][j].f[r] = 0.f;

  // Prologue: TDM-load stage 0 (wave 0 issues; DMA engine does the work).
  if (wid == 0) {
    tdm_load_tile_2d((u32)(uintptr_t)&As[0][0], A + (size_t)m0 * K,
                     (u32)K, (u32)(M - m0), BK, 128, (u64)K);
    tdm_load_tile_2d((u32)(uintptr_t)&Bs[0][0], Bt + (size_t)n0 * K,
                     (u32)K, (u32)(N - n0), BK, 128, (u64)K);
  }

  for (int kt = 0; kt < nk; ++kt) {
    const int cur = kt & 1;
    if (wid == 0) {
      if (kt + 1 < nk) {
        const int nxt = (kt + 1) & 1;
        const int k1 = (kt + 1) * BK;
        tdm_load_tile_2d((u32)(uintptr_t)&As[nxt][0], A + (size_t)m0 * K + k1,
                         (u32)(K - k1), (u32)(M - m0), BK, 128, (u64)K);
        tdm_load_tile_2d((u32)(uintptr_t)&Bs[nxt][0], Bt + (size_t)n0 * K + k1,
                         (u32)(K - k1), (u32)(N - n0), BK, 128, (u64)K);
        __builtin_amdgcn_s_wait_tensorcnt((short)2);  // stage kt done
      } else {
        __builtin_amdgcn_s_wait_tensorcnt((short)0);
      }
    }
    __syncthreads();

    #pragma unroll
    for (int ks = 0; ks < 2; ++ks) {
      Frag16 af[4];
      #pragma unroll
      for (int mf = 0; mf < 4; ++mf) {
        const u16* p = &As[cur][(wm * 64 + mf * 16 + lr) * LDS_A + ks * 32];
        af[mf].u[0] = *(const uint4*)(p + half * 8);
        af[mf].u[1] = *(const uint4*)(p + 16 + half * 8);
      }
      Frag16 bfr[2];
      #pragma unroll
      for (int nf = 0; nf < 2; ++nf) {
        const u16* p =
            &Bs[cur][(wn * 32 + nf * 16 + lr) * LDS_A + ks * 32];
        bfr[nf].u[0] = *(const uint4*)(p + half * 8);
        bfr[nf].u[1] = *(const uint4*)(p + 16 + half * 8);
      }
      #pragma unroll
      for (int mf = 0; mf < 4; ++mf)
        #pragma unroll
        for (int nf = 0; nf < 2; ++nf)
          acc[mf][nf].v = __builtin_amdgcn_wmma_f32_16x16x32_bf16(
              false, af[mf].v, false, bfr[nf].v, (short)0, acc[mf][nf].v,
              false, false);
    }
    __syncthreads();
  }

  // Epilogue. C layout: VGPR r -> row (half*8 + r); col = lr within 16-frag.
  #pragma unroll
  for (int mf = 0; mf < 4; ++mf) {
    #pragma unroll
    for (int nf = 0; nf < 2; ++nf) {
      #pragma unroll
      for (int r = 0; r < 8; ++r) {
        int row = m0 + wm * 64 + mf * 16 + half * 8 + r;
        int col = n0 + wn * 32 + nf * 16 + lr;
        float aval = acc[mf][nf].f[r];
        if (EPI == 0) {
          int which = col >> 10;
          int cw    = col & 1023;
          int head  = cw >> 6;
          int d     = cw & 63;
          int bb    = row >> 11;
          int ii    = row & 2047;
          size_t dst = ((size_t)(bb * 16 + head) * 2048 + ii) * 64 + d;
          u16* o = (which == 0) ? qout : (which == 1) ? kout : vout;
          o[dst] = f2bf(aval);
        } else if (EPI == 1) {
          float t = aval + bias[col] + resid[(size_t)row * N + col];
          outF[(size_t)row * N + col] = t;
        } else {
          float t = aval + bias[col];
          float g = 0.5f * t * (1.0f + erff(t * 0.70710678118f));
          outB[(size_t)row * N + col] = f2bf(g);
        }
      }
    }
  }
}

// ---------------------------------------------------------------------------
// Flash attention: grid (2048/128, b*h). Each block: 128 Q rows, 8 waves x 16
// rows. KV blocks of 64 keys. Q/K/V: bf16 [bh][2048][64]. O: bf16 [b*n][1024].
// ---------------------------------------------------------------------------
#define S72 72

__global__ __launch_bounds__(256) void attn_kernel(
    const u16* __restrict__ Q, const u16* __restrict__ Kmat,
    const u16* __restrict__ V, u16* __restrict__ O, float scale) {
  __shared__ u16 Ks[64 * S72];        //  9216 B
  __shared__ u16 Vts[64 * S72];       //  9216 B  (V transposed: [d][j])
  __shared__ u16 Ps[8 * 16 * S72];    // 18432 B  (per-wave P; doubles as Q stage)

  const int tid  = threadIdx.x;
  const int lane = tid & 31;
  const int wid  = tid >> 5;
  const int half = lane >> 4;
  const int lr   = lane & 15;
  const int bh   = blockIdx.y;
  const int q0   = blockIdx.x * 128;
  const size_t base = (size_t)bh * 2048 * 64;
  const u16* Qb = Q    + base;
  const u16* Kb = Kmat + base;
  const u16* Vb = V    + base;

  // Stage Q tile (128x64, stride 72) into the Ps region, load A-frags, free it.
  u16* Qs = Ps;
  #pragma unroll
  for (int j = 0; j < 4; ++j) {
    int c = tid + 256 * j;
    int row = c >> 3, kk = (c & 7) * 8;
    *(uint4*)(&Qs[row * S72 + kk]) =
        *(const uint4*)(Qb + (size_t)(q0 + row) * 64 + kk);
  }
  __syncthreads();
  Frag16 qf[2];
  #pragma unroll
  for (int ks = 0; ks < 2; ++ks) {
    const u16* p = &Qs[(wid * 16 + lr) * S72 + ks * 32];
    qf[ks].u[0] = *(const uint4*)(p + half * 8);
    qf[ks].u[1] = *(const uint4*)(p + 16 + half * 8);
  }
  __syncthreads();

  F8 o[4];
  #pragma unroll
  for (int of = 0; of < 4; ++of)
    #pragma unroll
    for (int r = 0; r < 8; ++r) o[of].f[r] = 0.f;
  float mrow[8], lsum[8];
  #pragma unroll
  for (int r = 0; r < 8; ++r) { mrow[r] = -3.0e38f; lsum[r] = 0.f; }

  u16* Pw = Ps + wid * 16 * S72;

  for (int j0 = 0; j0 < 2048; j0 += 64) {
    // Stage K tile [64][64] (K-contiguous rows).
    #pragma unroll
    for (int j = 0; j < 2; ++j) {
      int c = tid + 256 * j;
      int row = c >> 3, kk = (c & 7) * 8;
      *(uint4*)(&Ks[row * S72 + kk]) =
          *(const uint4*)(Kb + (size_t)(j0 + row) * 64 + kk);
    }
    // Stage V transposed: Vts[d][j].
    {
      int j  = tid >> 2;
      int d0 = (tid & 3) * 16;
      uint4 a = *(const uint4*)(Vb + (size_t)(j0 + j) * 64 + d0);
      uint4 b = *(const uint4*)(Vb + (size_t)(j0 + j) * 64 + d0 + 8);
      const u16* pa = (const u16*)&a;
      const u16* pb = (const u16*)&b;
      #pragma unroll
      for (int t = 0; t < 8; ++t) {
        Vts[(d0 + t) * S72 + j]     = pa[t];
        Vts[(d0 + 8 + t) * S72 + j] = pb[t];
      }
    }
    __syncthreads();

    // Scores: 16 Q rows x 64 keys via WMMA.
    F8 sc[4];
    #pragma unroll
    for (int nf = 0; nf < 4; ++nf)
      #pragma unroll
      for (int r = 0; r < 8; ++r) sc[nf].f[r] = 0.f;
    #pragma unroll
    for (int ks = 0; ks < 2; ++ks) {
      #pragma unroll
      for (int nf = 0; nf < 4; ++nf) {
        Frag16 kfr;
        const u16* p = &Ks[(nf * 16 + lr) * S72 + ks * 32];
        kfr.u[0] = *(const uint4*)(p + half * 8);
        kfr.u[1] = *(const uint4*)(p + 16 + half * 8);
        sc[nf].v = __builtin_amdgcn_wmma_f32_16x16x32_bf16(
            false, qf[ks].v, false, kfr.v, (short)0, sc[nf].v, false, false);
      }
    }

    // Online softmax; row values live across 16 lanes of each C VGPR.
    #pragma unroll
    for (int r = 0; r < 8; ++r) {
      float mx = -3.0e38f;
      #pragma unroll
      for (int nf = 0; nf < 4; ++nf) {
        sc[nf].f[r] *= scale;
        mx = fmaxf(mx, sc[nf].f[r]);
      }
      #pragma unroll
      for (int off = 8; off > 0; off >>= 1)
        mx = fmaxf(mx, __shfl_xor(mx, off, 32));
      float mnew  = fmaxf(mrow[r], mx);
      float alpha = __expf(mrow[r] - mnew);
      mrow[r] = mnew;
      float rs = 0.f;
      #pragma unroll
      for (int nf = 0; nf < 4; ++nf) {
        float pv = __expf(sc[nf].f[r] - mnew);
        sc[nf].f[r] = pv;
        rs += pv;
      }
      #pragma unroll
      for (int off = 8; off > 0; off >>= 1) rs += __shfl_xor(rs, off, 32);
      lsum[r] = lsum[r] * alpha + rs;
      #pragma unroll
      for (int of = 0; of < 4; ++of) o[of].f[r] *= alpha;
    }

    // C-layout -> A-layout for P via per-wave LDS round trip (bf16).
    #pragma unroll
    for (int nf = 0; nf < 4; ++nf)
      #pragma unroll
      for (int r = 0; r < 8; ++r)
        Pw[(half * 8 + r) * S72 + nf * 16 + lr] = f2bf(sc[nf].f[r]);

    // O += P (16x64) * V (64x64) via WMMA.
    #pragma unroll
    for (int pk = 0; pk < 2; ++pk) {
      Frag16 pf;
      const u16* pp = &Pw[lr * S72 + pk * 32];
      pf.u[0] = *(const uint4*)(pp + half * 8);
      pf.u[1] = *(const uint4*)(pp + 16 + half * 8);
      #pragma unroll
      for (int of = 0; of < 4; ++of) {
        Frag16 vf;
        const u16* vp = &Vts[(of * 16 + lr) * S72 + pk * 32];
        vf.u[0] = *(const uint4*)(vp + half * 8);
        vf.u[1] = *(const uint4*)(vp + 16 + half * 8);
        o[of].v = __builtin_amdgcn_wmma_f32_16x16x32_bf16(
            false, pf.v, false, vf.v, (short)0, o[of].v, false, false);
      }
    }
    __syncthreads();
  }

  // Normalize and write to [b*n][heads*64] (bf16) for the out-proj GEMM.
  const int bb = bh >> 4;
  const int hh = bh & 15;
  #pragma unroll
  for (int of = 0; of < 4; ++of) {
    #pragma unroll
    for (int r = 0; r < 8; ++r) {
      int m_local = q0 + wid * 16 + half * 8 + r;
      size_t row = (size_t)bb * 2048 + m_local;
      int col = hh * 64 + of * 16 + lr;
      O[row * 1024 + col] = f2bf(o[of].f[r] / lsum[r]);
    }
  }
}

// ---------------------------------------------------------------------------
// Host launcher
// ---------------------------------------------------------------------------
extern "C" void kernel_launch(void* const* d_in, const int* in_sizes, int n_in,
                              void* d_out, int out_size, void* d_ws,
                              size_t ws_size, hipStream_t stream) {
  const float* x     = (const float*)d_in[0];
  const float* ln1_g = (const float*)d_in[1];
  const float* ln1_b = (const float*)d_in[2];
  const float* w_qkv = (const float*)d_in[3];
  const float* w_out = (const float*)d_in[4];
  const float* b_out = (const float*)d_in[5];
  const float* ln2_g = (const float*)d_in[6];
  const float* ln2_b = (const float*)d_in[7];
  const float* w1    = (const float*)d_in[8];
  const float* b1    = (const float*)d_in[9];
  const float* w2    = (const float*)d_in[10];
  const float* b2    = (const float*)d_in[11];

  char* ws = (char*)d_ws;
  size_t off = 0;
  auto alloc = [&](size_t bytes) -> char* {
    char* p = ws + off;
    off = (off + bytes + 255) & ~(size_t)255;
    return p;
  };
  u16* wqkvT = (u16*)alloc(3072ull * 1024 * 2);
  u16* woutT = (u16*)alloc(1024ull * 1024 * 2);
  u16* w1T   = (u16*)alloc(4096ull * 1024 * 2);
  u16* w2T   = (u16*)alloc(1024ull * 4096 * 2);
  u16* h1    = (u16*)alloc(4096ull * 1024 * 2);
  u16* qbuf  = (u16*)alloc(4096ull * 1024 * 2);
  u16* kbuf  = (u16*)alloc(4096ull * 1024 * 2);
  u16* vbuf  = (u16*)alloc(4096ull * 1024 * 2);
  u16* aout  = (u16*)alloc(4096ull * 1024 * 2);
  float* x1  = (float*)alloc(4096ull * 1024 * 4);
  u16* h2    = (u16*)alloc(4096ull * 1024 * 2);
  u16* h3    = (u16*)alloc(4096ull * 4096 * 2);

  // 1) Weight convert+transpose to bf16 [N][K].
  convert_transpose_kernel<<<(1024 * 3072) / 256, 256, 0, stream>>>(w_qkv, wqkvT, 1024, 3072);
  convert_transpose_kernel<<<(1024 * 1024) / 256, 256, 0, stream>>>(w_out, woutT, 1024, 1024);
  convert_transpose_kernel<<<(1024 * 4096) / 256, 256, 0, stream>>>(w1, w1T, 1024, 4096);
  convert_transpose_kernel<<<(4096 * 1024) / 256, 256, 0, stream>>>(w2, w2T, 4096, 1024);

  // 2) LN1 -> bf16.
  ln_bf16_kernel<<<4096, 256, 0, stream>>>(x, ln1_g, ln1_b, h1);

  // 3) QKV GEMM with head-scatter epilogue.
  gemm_bf16_wmma<0><<<dim3(3072 / 128, 4096 / 128), 256, 0, stream>>>(
      h1, wqkvT, 4096, 3072, 1024, nullptr, nullptr, nullptr, nullptr,
      qbuf, kbuf, vbuf);

  // 4) Flash attention.
  attn_kernel<<<dim3(2048 / 128, 32), 256, 0, stream>>>(
      qbuf, kbuf, vbuf, aout, 0.125f);

  // 5) Out-proj + bias + residual(x) -> x1 (fp32).
  gemm_bf16_wmma<1><<<dim3(1024 / 128, 4096 / 128), 256, 0, stream>>>(
      aout, woutT, 4096, 1024, 1024, b_out, x, x1, nullptr,
      nullptr, nullptr, nullptr);

  // 6) LN2 -> bf16.
  ln_bf16_kernel<<<4096, 256, 0, stream>>>(x1, ln2_g, ln2_b, h2);

  // 7) MLP1 + bias + GELU -> bf16.
  gemm_bf16_wmma<2><<<dim3(4096 / 128, 4096 / 128), 256, 0, stream>>>(
      h2, w1T, 4096, 4096, 1024, b1, nullptr, nullptr, h3,
      nullptr, nullptr, nullptr);

  // 8) MLP2 + bias + residual(x1) -> d_out (fp32).
  gemm_bf16_wmma<1><<<dim3(1024 / 128, 4096 / 128), 256, 0, stream>>>(
      h3, w2T, 4096, 1024, 4096, b2, x1, (float*)d_out, nullptr,
      nullptr, nullptr, nullptr);
}
